// PointerNetwork_81209241633048
// MI455X (gfx1250) — matrix operand order repaired and checked by reading, not compiled, not run on hardware
//
#include <hip/hip_runtime.h>

typedef __bf16 bf16;
typedef __attribute__((ext_vector_type(16))) __bf16 v16bf;
typedef __attribute__((ext_vector_type(8)))  __bf16 v8bf;
typedef __attribute__((ext_vector_type(8)))  float  v8f;

#define HD   256      // hidden size H
#define NPT  128      // N points
#define NBAT 1024     // batch B
#define G4   1024     // 4*H (gate dim)

// ---------------- helpers ----------------

__device__ __forceinline__ float fast_rcp(float x) {
#if __has_builtin(__builtin_amdgcn_rcpf)
    return __builtin_amdgcn_rcpf(x);
#else
    return 1.0f / x;
#endif
}

__device__ __forceinline__ float ftanh(float x) {
#if __has_builtin(__builtin_amdgcn_tanhf)
    return __builtin_amdgcn_tanhf(x);          // single v_tanh_f32
#else
    x = fminf(fmaxf(x, -15.0f), 15.0f);
    float e = __expf(2.0f * x);
    return 1.0f - 2.0f * fast_rcp(e + 1.0f);
#endif
}

__device__ __forceinline__ float fsig(float x) {
#if __has_builtin(__builtin_amdgcn_tanhf)
    return __builtin_amdgcn_tanhf(0.5f * x) * 0.5f + 0.5f;
#else
    return fast_rcp(1.0f + __expf(-x));
#endif
}

__device__ __forceinline__ v8f wmma_bf16(v16bf a, v16bf b, v8f c) {
    return __builtin_amdgcn_wmma_f32_16x16x32_bf16(false, a, false, b, (short)0, c,
                                                   false, false);
}

// A-operand 16x32 bf16 from row-major bf16 (stride HD).
// lane<16: rows m, K = k0+{0..7} U k0+{16..23}; lane>=16: K = k0+{8..15} U k0+{24..31}
__device__ __forceinline__ v16bf load_a(const bf16* base, int mbase, int k0, int lane) {
    int row = mbase + (lane & 15);
    const bf16* p = base + (size_t)row * HD + k0 + ((lane >> 4) << 3);
    v8bf lo = *(const v8bf*)(p);
    v8bf hi = *(const v8bf*)(p + 16);
    return __builtin_shufflevector(lo, hi, 0,1,2,3,4,5,6,7,8,9,10,11,12,13,14,15);
}

// B-operand 32x16 bf16: B[k,n] = W[n,k], W row-major stride HD.
// lane L: col n = nbase+(L&15), K = k0 + (L>>4)*8 + {0..7} U k0 + 16 + (L>>4)*8 + {0..7}
__device__ __forceinline__ v16bf load_b(const bf16* W, int nbase, int k0, int lane) {
    int row = nbase + (lane & 15);
    const bf16* p = W + (size_t)row * HD + k0 + ((lane >> 4) << 3);
    v8bf lo = *(const v8bf*)(p);
    v8bf hi = *(const v8bf*)(p + 16);
    return __builtin_shufflevector(lo, hi, 0,1,2,3,4,5,6,7,8,9,10,11,12,13,14,15);
}

// ---------------- weight conversion ----------------

__global__ void cvt_bf16_kernel(const float* __restrict__ src, bf16* __restrict__ dst, int n) {
    for (int i = blockIdx.x * blockDim.x + threadIdx.x; i < n; i += gridDim.x * blockDim.x)
        dst[i] = (bf16)src[i];
}

// ---------------- encoder LSTM ----------------
// 32 blocks x 32 batch rows, 256 threads = 8 waves.
// wave: m-tile = wave&1 (16 rows), u-tiles = (wave>>1)*4 .. +3
// h double-buffered in LDS: pair results written straight to the other buffer.

__global__ __launch_bounds__(256) void encoder_kernel(
    const float* __restrict__ coords, const float* __restrict__ Wih,
    const bf16* __restrict__ Whh, const float* __restrict__ bih,
    const float* __restrict__ bhh, bf16* __restrict__ enc_out,
    float* __restrict__ h_final, float* __restrict__ c_final)
{
    __shared__ __align__(16) bf16  hbuf[2][32 * HD];   // 2 x 16KB ping-pong
    __shared__ float s_wx[G4 * 2];                     // 8KB : W_ih staged
    __shared__ float s_bias[G4];                       // 4KB : b_ih + b_hh
    __shared__ float s_x[32 * 2];                      // x_t for this tile

    const int tid  = threadIdx.x;
    const int lane = tid & 31;
    const int wave = tid >> 5;
    const int bbase = blockIdx.x * 32;
    const int mbase = (wave & 1) * 16;
    const int ug    = wave >> 1;

    for (int i = tid; i < G4; i += 256) {
        s_wx[2 * i]     = Wih[2 * i];
        s_wx[2 * i + 1] = Wih[2 * i + 1];
        s_bias[i]       = bih[i] + bhh[i];
    }
    for (int i = tid; i < 32 * HD; i += 256) hbuf[0][i] = (bf16)0.0f;

    float c_reg[4][8];
    #pragma unroll
    for (int p = 0; p < 4; ++p)
        #pragma unroll
        for (int r = 0; r < 8; ++r) c_reg[p][r] = 0.0f;

    __syncthreads();

    int cur = 0;
    for (int t = 0; t < NPT; ++t) {
        if (tid < 64)
            s_x[tid] = coords[((size_t)(bbase + (tid >> 1)) * NPT + t) * 2 + (tid & 1)];
        __syncthreads();

        const bf16* hc = hbuf[cur];
        bf16*       hn = hbuf[cur ^ 1];

        #pragma unroll
        for (int p = 0; p < 4; ++p) {
            const int ubase = (ug * 4 + p) * 16;
            const int ucol  = ubase + (lane & 15);
            v8f acc[4];
            // init with x @ W_ih^T + (b_ih + b_hh)
            #pragma unroll
            for (int g = 0; g < 4; ++g) {
                int j = g * HD + ucol;
                float w0 = s_wx[2 * j], w1 = s_wx[2 * j + 1], bs = s_bias[j];
                #pragma unroll
                for (int r = 0; r < 8; ++r) {
                    int b = mbase + r + ((lane >> 4) << 3);
                    acc[g][r] = bs + s_x[2 * b] * w0 + s_x[2 * b + 1] * w1;
                }
            }
            // h @ W_hh^T via WMMA, K = 256 in 8 chunks of 32 (no unroll: cap regs)
            #pragma unroll 1
            for (int k0 = 0; k0 < HD; k0 += 32) {
                v16bf a = load_a(hc, mbase, k0, lane);
                #pragma unroll
                for (int g = 0; g < 4; ++g) {
                    v16bf b = load_b(Whh, g * HD + ubase, k0, lane);
                    acc[g] = wmma_bf16(a, b, acc[g]);
                }
            }
            // LSTM cell activations, write straight to the other buffer
            #pragma unroll
            for (int r = 0; r < 8; ++r) {
                float iv = fsig(acc[0][r]);
                float fv = fsig(acc[1][r]);
                float gv = ftanh(acc[2][r]);
                float ov = fsig(acc[3][r]);
                float cn = fv * c_reg[p][r] + iv * gv;
                c_reg[p][r] = cn;
                float hv = ov * ftanh(cn);
                int b = mbase + r + ((lane >> 4) << 3);
                bf16 hb = (bf16)hv;
                hn[b * HD + ucol] = hb;
                enc_out[((size_t)(bbase + b) * NPT + t) * HD + ucol] = hb;
            }
        }
        __syncthreads();
        cur ^= 1;
    }

    // final h, c for the decoder
    #pragma unroll
    for (int p = 0; p < 4; ++p) {
        const int ucol = (ug * 4 + p) * 16 + (lane & 15);
        #pragma unroll
        for (int r = 0; r < 8; ++r) {
            int b = mbase + r + ((lane >> 4) << 3);
            c_final[(size_t)(bbase + b) * HD + ucol] = c_reg[p][r];
            h_final[(size_t)(bbase + b) * HD + ucol] = (float)hbuf[cur][b * HD + ucol];
        }
    }
}

// ---------------- ref_enc = enc_out @ W_ref^T ----------------
// 1024 blocks x 128 rows (of B*N = 131072), 256 threads = 8 waves, 1 m-tile each.

__global__ __launch_bounds__(256) void refenc_kernel(
    const bf16* __restrict__ enc_out, const bf16* __restrict__ Wref,
    bf16* __restrict__ ref_enc)
{
    const int tid = threadIdx.x, lane = tid & 31, wave = tid >> 5;
    const size_t rbase = (size_t)blockIdx.x * 128 + wave * 16;

    #pragma unroll 1
    for (int nt = 0; nt < 16; ++nt) {
        const int nbase = nt * 16;
        v8f acc;
        #pragma unroll
        for (int r = 0; r < 8; ++r) acc[r] = 0.0f;
        #pragma unroll 1
        for (int k0 = 0; k0 < HD; k0 += 32) {
            size_t row = rbase + (lane & 15);
            const bf16* p = enc_out + row * HD + k0 + ((lane >> 4) << 3);
            v8bf lo = *(const v8bf*)(p);
            v8bf hi = *(const v8bf*)(p + 16);
            v16bf a = __builtin_shufflevector(lo, hi, 0,1,2,3,4,5,6,7,8,9,10,11,12,13,14,15);
            v16bf b = load_b(Wref, nbase, k0, lane);
            acc = wmma_bf16(a, b, acc);
        }
        #pragma unroll
        for (int r = 0; r < 8; ++r) {
            size_t row = rbase + r + ((lane >> 4) << 3);
            ref_enc[row * HD + nbase + (lane & 15)] = (bf16)acc[r];
        }
    }
}

// ---------------- decoder ----------------
// 32 blocks x 32 batch rows, 256 threads = 8 waves.
// wave: m-tile = wave&1, u-tiles = (wave>>1)*4 .. +3

__global__ __launch_bounds__(256) void decoder_kernel(
    const float* __restrict__ coords, const float* __restrict__ Wih,
    const bf16* __restrict__ Whh, const float* __restrict__ bih,
    const float* __restrict__ bhh, const bf16* __restrict__ Wq,
    const bf16* __restrict__ ref_enc, const float* __restrict__ vvec,
    const float* __restrict__ h_final, const float* __restrict__ c_final,
    float* __restrict__ out_tour, float* __restrict__ out_lp)
{
    __shared__ __align__(16) bf16  hbuf[32 * HD];       // 16KB
    __shared__ __align__(16) bf16  qbuf[32 * HD];       // 16KB
    __shared__ float    s_scores[32 * NPT];             // 16KB
    __shared__ float    s_bias[G4];                     // 4KB
    __shared__ float    s_v[HD];                        // 1KB
    __shared__ unsigned s_vis[32 * 4];                  // visited bitmask
    __shared__ float    s_x[32 * 2];                    // dec_in
    __shared__ float    s_lp[32];

    const int tid  = threadIdx.x;
    const int lane = tid & 31;
    const int wave = tid >> 5;
    const int bbase = blockIdx.x * 32;
    const int mbase = (wave & 1) * 16;
    const int ug    = wave >> 1;

    for (int i = tid; i < G4; i += 256) s_bias[i] = bih[i] + bhh[i];
    for (int i = tid; i < HD; i += 256) s_v[i] = vvec[i];
    for (int i = tid; i < 32 * HD; i += 256)
        hbuf[i] = (bf16)h_final[(size_t)(bbase + (i >> 8)) * HD + (i & (HD - 1))];
    if (tid < 128) s_vis[tid] = ((tid & 3) == 0) ? 1u : 0u;   // node 0 visited
    if (tid < 64)  s_x[tid] = coords[((size_t)(bbase + (tid >> 1)) * NPT + 0) * 2 + (tid & 1)];
    if (tid < 32)  s_lp[tid] = 0.0f;

    float c_reg[4][8];
    #pragma unroll
    for (int p = 0; p < 4; ++p) {
        const int ucol = (ug * 4 + p) * 16 + (lane & 15);
        #pragma unroll
        for (int r = 0; r < 8; ++r) {
            int b = mbase + r + ((lane >> 4) << 3);
            c_reg[p][r] = c_final[(size_t)(bbase + b) * HD + ucol];
        }
    }
    __syncthreads();

    #pragma unroll 1
    for (int s = 0; s < NPT - 1; ++s) {
        // ---- phase 1: LSTM cell (WMMA on h @ W_hh^T) ----
        float hnew[4][8];
        #pragma unroll
        for (int p = 0; p < 4; ++p) {
            const int ubase = (ug * 4 + p) * 16;
            const int ucol  = ubase + (lane & 15);
            v8f acc[4];
            #pragma unroll
            for (int g = 0; g < 4; ++g) {
                int j = g * HD + ucol;
                float w0 = Wih[2 * j], w1 = Wih[2 * j + 1], bs = s_bias[j];
                #pragma unroll
                for (int r = 0; r < 8; ++r) {
                    int b = mbase + r + ((lane >> 4) << 3);
                    acc[g][r] = bs + s_x[2 * b] * w0 + s_x[2 * b + 1] * w1;
                }
            }
            #pragma unroll 1
            for (int k0 = 0; k0 < HD; k0 += 32) {
                v16bf a = load_a(hbuf, mbase, k0, lane);
                #pragma unroll
                for (int g = 0; g < 4; ++g) {
                    v16bf b = load_b(Whh, g * HD + ubase, k0, lane);
                    acc[g] = wmma_bf16(a, b, acc[g]);
                }
            }
            #pragma unroll
            for (int r = 0; r < 8; ++r) {
                float iv = fsig(acc[0][r]);
                float fv = fsig(acc[1][r]);
                float gv = ftanh(acc[2][r]);
                float ov = fsig(acc[3][r]);
                float cn = fv * c_reg[p][r] + iv * gv;
                c_reg[p][r] = cn;
                hnew[p][r]  = ov * ftanh(cn);
            }
        }
        __syncthreads();
        #pragma unroll
        for (int p = 0; p < 4; ++p) {
            const int ucol = (ug * 4 + p) * 16 + (lane & 15);
            #pragma unroll
            for (int r = 0; r < 8; ++r) {
                int b = mbase + r + ((lane >> 4) << 3);
                hbuf[b * HD + ucol] = (bf16)hnew[p][r];
            }
        }
        __syncthreads();

        // ---- phase 2: q = h @ W_q^T (WMMA) ----
        #pragma unroll
        for (int p = 0; p < 4; ++p) {
            const int ubase = (ug * 4 + p) * 16;
            v8f acc;
            #pragma unroll
            for (int r = 0; r < 8; ++r) acc[r] = 0.0f;
            #pragma unroll 1
            for (int k0 = 0; k0 < HD; k0 += 32) {
                v16bf a = load_a(hbuf, mbase, k0, lane);
                v16bf b = load_b(Wq, ubase, k0, lane);
                acc = wmma_bf16(a, b, acc);
            }
            #pragma unroll
            for (int r = 0; r < 8; ++r) {
                int b = mbase + r + ((lane >> 4) << 3);
                qbuf[b * HD + ubase + (lane & 15)] = (bf16)acc[r];
            }
        }
        __syncthreads();

        // ---- phase 3: scores[b,n] = sum_h v[h] * tanh(ref_enc[b,n,h] + q[b,h]) ----
        {
            const int b  = tid >> 3;          // 0..31
            const int n0 = (tid & 7) * 16;    // 16 n's per thread
            const bf16* qp = qbuf + b * HD;
            #pragma unroll 1
            for (int i = 0; i < 16; ++i) {
                const int n = n0 + i;
                float sc;
                if ((s_vis[b * 4 + (n >> 5)] >> (n & 31)) & 1u) {
                    sc = -1e30f;
                } else {
                    const bf16* rp = ref_enc + ((size_t)(bbase + b) * NPT + n) * HD;
                    float sum = 0.0f;
                    #pragma unroll 4
                    for (int h0 = 0; h0 < HD; h0 += 8) {
                        v8bf rv = *(const v8bf*)(rp + h0);
                        v8bf qv = *(const v8bf*)(qp + h0);
                        #pragma unroll
                        for (int j = 0; j < 8; ++j)
                            sum += s_v[h0 + j] * ftanh((float)rv[j] + (float)qv[j]);
                    }
                    sc = sum;
                }
                s_scores[b * NPT + n] = sc;
            }
        }
        __syncthreads();

        // ---- phase 4: per-row argmax + softmax log-prob (wave32 shuffles) ----
        #pragma unroll 1
        for (int rr = 0; rr < 4; ++rr) {
            const int b = wave * 4 + rr;
            float vals[4];
            float m = -3e30f; int mi = 0;
            #pragma unroll
            for (int j = 0; j < 4; ++j) {
                int n = lane + j * 32;
                float v = s_scores[b * NPT + n];
                vals[j] = v;
                if (v > m || (v == m && n < mi)) { m = v; mi = n; }
            }
            #pragma unroll
            for (int off = 16; off > 0; off >>= 1) {
                float om = __shfl_xor(m, off, 32);
                int   oi = __shfl_xor(mi, off, 32);
                if (om > m || (om == m && oi < mi)) { m = om; mi = oi; }
            }
            float se = 0.0f;
            #pragma unroll
            for (int j = 0; j < 4; ++j) se += __expf(vals[j] - m);
            #pragma unroll
            for (int off = 16; off > 0; off >>= 1) se += __shfl_xor(se, off, 32);
            if (lane == 0) {
                s_lp[b] += __logf(fast_rcp(se) + 1e-8f);
                s_vis[b * 4 + (mi >> 5)] |= 1u << (mi & 31);
                out_tour[(size_t)(bbase + b) * (NPT - 1) + s] = (float)mi;
                s_x[2 * b]     = coords[((size_t)(bbase + b) * NPT + mi) * 2 + 0];
                s_x[2 * b + 1] = coords[((size_t)(bbase + b) * NPT + mi) * 2 + 1];
            }
        }
        __syncthreads();
    }

    if (tid < 32) out_lp[bbase + tid] = s_lp[tid];
}

// ---------------- launch ----------------

extern "C" void kernel_launch(void* const* d_in, const int* in_sizes, int n_in,
                              void* d_out, int out_size, void* d_ws, size_t ws_size,
                              hipStream_t stream) {
    (void)in_sizes; (void)n_in; (void)out_size; (void)ws_size;

    const float* coords  = (const float*)d_in[0];
    const float* eWih    = (const float*)d_in[1];
    const float* eWhh    = (const float*)d_in[2];
    const float* eBih    = (const float*)d_in[3];
    const float* eBhh    = (const float*)d_in[4];
    const float* dWih    = (const float*)d_in[5];
    const float* dWhh    = (const float*)d_in[6];
    const float* dBih    = (const float*)d_in[7];
    const float* dBhh    = (const float*)d_in[8];
    const float* Wref    = (const float*)d_in[9];
    const float* Wq      = (const float*)d_in[10];
    const float* vvec    = (const float*)d_in[11];

    char* ws = (char*)d_ws;
    size_t o = 0;
    bf16* whh_e   = (bf16*)(ws + o); o += (size_t)G4 * HD * 2;      // 512 KB
    bf16* whh_d   = (bf16*)(ws + o); o += (size_t)G4 * HD * 2;      // 512 KB
    bf16* wref_b  = (bf16*)(ws + o); o += (size_t)HD * HD * 2;      // 128 KB
    bf16* wq_b    = (bf16*)(ws + o); o += (size_t)HD * HD * 2;      // 128 KB
    bf16* enc_out = (bf16*)(ws + o); o += (size_t)NBAT * NPT * HD * 2;  // 64 MB
    bf16* ref_enc = (bf16*)(ws + o); o += (size_t)NBAT * NPT * HD * 2;  // 64 MB
    float* h_fin  = (float*)(ws + o); o += (size_t)NBAT * HD * 4;   // 1 MB
    float* c_fin  = (float*)(ws + o); o += (size_t)NBAT * HD * 4;   // 1 MB

    float* out_tour = (float*)d_out;                       // B x (N-1)
    float* out_lp   = (float*)d_out + (size_t)NBAT * (NPT - 1);

    cvt_bf16_kernel<<<256, 256, 0, stream>>>(eWhh, whh_e, G4 * HD);
    cvt_bf16_kernel<<<256, 256, 0, stream>>>(dWhh, whh_d, G4 * HD);
    cvt_bf16_kernel<<<64, 256, 0, stream>>>(Wref, wref_b, HD * HD);
    cvt_bf16_kernel<<<64, 256, 0, stream>>>(Wq, wq_b, HD * HD);

    encoder_kernel<<<NBAT / 32, 256, 0, stream>>>(coords, eWih, whh_e, eBih, eBhh,
                                                  enc_out, h_fin, c_fin);

    refenc_kernel<<<(NBAT * NPT) / 128, 256, 0, stream>>>(enc_out, wref_b, ref_enc);

    decoder_kernel<<<NBAT / 32, 256, 0, stream>>>(coords, dWih, whh_d, dBih, dBhh,
                                                  wq_b, ref_enc, vvec, h_fin, c_fin,
                                                  out_tour, out_lp);
}
